// SCNWrapper_49881750176304
// MI455X (gfx1250) — matrix operand order repaired and checked by reading, not compiled, not run on hardware
//
#include <hip/hip_runtime.h>
#include <hip/hip_bf16.h>

typedef __bf16 bf16;
typedef __attribute__((ext_vector_type(16))) __bf16 v16bf;
typedef __attribute__((ext_vector_type(8)))  __bf16 v8bf;
typedef __attribute__((ext_vector_type(8)))  float  v8f;

#define C_DIM 128

// ---------------------------------------------------------------------------
// inv[r] = d!=0 ? rsqrt(d) : 0,  d = sum_c |L[r,c]|
// ---------------------------------------------------------------------------
__global__ void rowsum_inv_kernel(const float* __restrict__ L,
                                  float* __restrict__ inv, int M) {
  __shared__ float red[256];
  int r = blockIdx.x;
  int t = threadIdx.x;
  const float* row = L + (size_t)r * M;
  float s = 0.f;
  for (int c = t * 4; c < M; c += 256 * 4) {
    float4 v = *(const float4*)(row + c);
    s += fabsf(v.x) + fabsf(v.y) + fabsf(v.z) + fabsf(v.w);
  }
  red[t] = s;
  __syncthreads();
  for (int off = 128; off > 0; off >>= 1) {
    if (t < off) red[t] += red[t + off];
    __syncthreads();
  }
  if (t == 0) {
    float d = red[0];
    inv[r] = (d != 0.f) ? rsqrtf(d) : 0.f;
  }
}

// ---------------------------------------------------------------------------
// Yt[c][n] = (sum_k X[n][k]*W[k][c]) * (scale ? scale[n] : 1), bf16, TRANSPOSED
// block = 256 threads, 32 rows of X per block
// ---------------------------------------------------------------------------
__global__ void __launch_bounds__(256)
small_gemm_kernel(const float* __restrict__ X, const float* __restrict__ W,
                  const float* __restrict__ scale, bf16* __restrict__ Yt, int N) {
  __shared__ float sX[32][36];
  __shared__ float sW[32][132];
  int t = threadIdx.x;
  int tx = t & 31;                   // row within tile
  int ty = t >> 5;                   // column group 0..7
  int rowbase = blockIdx.x * 32;
  float acc[16];
#pragma unroll
  for (int i = 0; i < 16; ++i) acc[i] = 0.f;

  for (int kc = 0; kc < C_DIM; kc += 32) {
#pragma unroll
    for (int i = 0; i < 4; ++i) {            // W chunk 32x128
      int idx = t + i * 256;
      int kk = idx >> 5;
      int c4 = idx & 31;
      float4 v = *(const float4*)(W + (size_t)(kc + kk) * C_DIM + c4 * 4);
      sW[kk][c4 * 4 + 0] = v.x; sW[kk][c4 * 4 + 1] = v.y;
      sW[kk][c4 * 4 + 2] = v.z; sW[kk][c4 * 4 + 3] = v.w;
    }
    {                                         // X chunk 32x32
      int rr = t >> 3;
      int c4 = t & 7;
      float4 v = *(const float4*)(X + (size_t)(rowbase + rr) * C_DIM + kc + c4 * 4);
      sX[rr][c4 * 4 + 0] = v.x; sX[rr][c4 * 4 + 1] = v.y;
      sX[rr][c4 * 4 + 2] = v.z; sX[rr][c4 * 4 + 3] = v.w;
    }
    __syncthreads();
#pragma unroll
    for (int kk = 0; kk < 32; ++kk) {
      float xv = sX[tx][kk];
#pragma unroll
      for (int c = 0; c < 16; ++c)
        acc[c] = fmaf(xv, sW[kk][ty * 16 + c], acc[c]);
    }
    __syncthreads();
  }
  int n = rowbase + tx;
  float s = scale ? scale[n] : 1.f;
#pragma unroll
  for (int c = 0; c < 16; ++c)
    Yt[(size_t)(ty * 16 + c) * N + n] = (bf16)(acc[c] * s);
}

// ---------------------------------------------------------------------------
// out[M x 128] = post( rowscale ⊙ (A[MxK,f32] @ Bt^T[Kx128,bf16]) + addend )
// Bt stored transposed [128][K] so B fragments are contiguous-K LDS reads.
// block = 256 thr = 8 waves; block tile 128x128; wave tile 16x128; K step 32.
// Double-buffered LDS. A tile staged via registers (needs f32->bf16 cvt);
// B tile copied global->LDS with GLOBAL_LOAD_ASYNC_TO_LDS_B128 (ASYNCcnt,
// zero VGPR footprint). s_wait_asynccnt before each barrier makes the barrier
// a collective completion point for the async copies.
// ---------------------------------------------------------------------------
__global__ void __launch_bounds__(256, 1)
big_gemm_wmma(const float* __restrict__ A, const bf16* __restrict__ Bt,
              const float* __restrict__ rowscale, const float* __restrict__ addend,
              float* __restrict__ out, int M, int K, int applySigmoid) {
  const int LDA = 48, LDB = 48;                  // bf16 elems; 96B rows
  const int SZA = 128 * LDA, SZB = 128 * LDB;    // one buffer, elems
  __shared__ __align__(32) bf16 sA[2 * SZA];
  __shared__ __align__(32) bf16 sB[2 * SZB];

  int t = threadIdx.x;
  int wave = t >> 5;
  int lane = t & 31;
  int laneRow = lane & 15;
  int laneHalf = lane >> 4;
  int rowblk = blockIdx.x * 128;

  v8f acc[8];
#pragma unroll
  for (int i = 0; i < 8; ++i)
#pragma unroll
    for (int j = 0; j < 8; ++j) acc[i][j] = 0.f;

  float4 aReg[4];

  // ---- A tile: issue global loads into registers (converted at commit)
  auto issueA = [&](int kt) {
#pragma unroll
    for (int i = 0; i < 4; ++i) {
      int idx = t + i * 256;
      int r = idx >> 3;
      int c4 = idx & 7;
      aReg[i] = *(const float4*)(A + (size_t)(rowblk + r) * K + kt + c4 * 4);
    }
  };
  // ---- B tile: async copy global -> LDS, no VGPR data path (ASYNCcnt)
  auto issueB = [&](int kt, int buf) {
    bf16* dB = sB + buf * SZB;
#pragma unroll
    for (int i = 0; i < 2; ++i) {
      int idx = t + i * 256;
      int n = idx >> 2;
      int c = idx & 3;
      const bf16* src = Bt + (size_t)n * K + kt + c * 8;
      unsigned dst = (unsigned)(size_t)&dB[n * LDB + c * 8];  // LDS byte address
      asm volatile("global_load_async_to_lds_b128 %0, %1, off"
                   :: "v"(dst), "v"(src) : "memory");
    }
  };
  // ---- convert + store staged A registers into LDS buffer `buf`
  auto commitA = [&](int buf) {
    bf16* dA = sA + buf * SZA;
#pragma unroll
    for (int i = 0; i < 4; ++i) {
      int idx = t + i * 256;
      int r = idx >> 3;
      int c4 = idx & 7;
      bf16* d = &dA[r * LDA + c4 * 4];
      d[0] = (bf16)aReg[i].x; d[1] = (bf16)aReg[i].y;
      d[2] = (bf16)aReg[i].z; d[3] = (bf16)aReg[i].w;
    }
  };

  int nk = K / 32;
  issueA(0);
  issueB(0, 0);
  commitA(0);

  for (int it = 0; it < nk; ++it) {
    int cur = it & 1;
    asm volatile("s_wait_asynccnt 0x0" ::: "memory");  // B copies landed
    __syncthreads();
    bool more = (it + 1) < nk;
    if (more) {
      issueA((it + 1) * 32);                 // A loads in flight during WMMAs
      issueB((it + 1) * 32, cur ^ 1);        // B async copies in flight
    }
    if (it + 2 < nk) {                       // warm L2 for tile after next
      int r = t >> 1;
      int half = t & 1;
      __builtin_prefetch(A + (size_t)(rowblk + r) * K + (it + 2) * 32 + half * 16, 0, 2);
    }

    const bf16* sAc = sA + cur * SZA;
    const bf16* sBc = sB + cur * SZB;

    // A fragment (16x32 bf16): lanes 0-15 K 0..7/16..23, lanes 16-31 +8
    const bf16* pa = sAc + (wave * 16 + laneRow) * LDA + laneHalf * 8;
    v8bf alo = *(const v8bf*)pa;
    v8bf ahi = *(const v8bf*)(pa + 16);
    v16bf afrag;
#pragma unroll
    for (int j = 0; j < 8; ++j) { afrag[j] = alo[j]; afrag[j + 8] = ahi[j]; }

    // 8 column tiles: B fragment from LDS, then WMMA (A fragment reused)
    const bf16* pb = sBc + laneRow * LDB + laneHalf * 8;
#pragma unroll
    for (int tc = 0; tc < 8; ++tc) {
      const bf16* pt = pb + tc * 16 * LDB;
      v8bf b0 = *(const v8bf*)pt;
      v8bf b1 = *(const v8bf*)(pt + 16);
      v16bf bfrag;
#pragma unroll
      for (int j = 0; j < 8; ++j) { bfrag[j] = b0[j]; bfrag[j + 8] = b1[j]; }
      acc[tc] = __builtin_amdgcn_wmma_f32_16x16x32_bf16(
          false, afrag, false, bfrag, (short)0, acc[tc], false, false);
    }

    if (more) commitA(cur ^ 1);              // s_wait_loadcnt lands here
  }

  // epilogue: D layout VGPR r -> M = r + 8*laneHalf, N = laneRow
#pragma unroll
  for (int r = 0; r < 8; ++r) {
    int grow = rowblk + wave * 16 + r + laneHalf * 8;
    float rs = rowscale ? rowscale[grow] : 1.f;
#pragma unroll
    for (int tc = 0; tc < 8; ++tc) {
      int gcol = tc * 16 + laneRow;
      float v = acc[tc][r] * rs;
      if (addend) v += addend[(size_t)grow * C_DIM + gcol];
      if (applySigmoid) v = 1.f / (1.f + __expf(-v));
      out[(size_t)grow * C_DIM + gcol] = v;
    }
  }
}

// ---------------------------------------------------------------------------
// Graph norm (segment mean/var over G=64 groups), 4 light passes
// ---------------------------------------------------------------------------
__global__ void gn_zero_kernel(float* gsum, float* gvar, float* gcnt) {
  int i = blockIdx.x * blockDim.x + threadIdx.x;
  if (i < 64 * C_DIM) { gsum[i] = 0.f; gvar[i] = 0.f; }
  if (i < 64) gcnt[i] = 0.f;
}
__global__ void gn_sum_kernel(const float* __restrict__ x, const int* __restrict__ batch,
                              float* gsum, float* gcnt, int N) {
  int i = blockIdx.x * blockDim.x + threadIdx.x;
  if (i >= N * C_DIM) return;
  int r = i / C_DIM, c = i % C_DIM;
  int g = batch[r];
  atomicAdd(&gsum[g * C_DIM + c], x[i]);
  if (c == 0) atomicAdd(&gcnt[g], 1.f);
}
__global__ void gn_center_kernel(float* __restrict__ x, const int* __restrict__ batch,
                                 const float* __restrict__ gsum, const float* __restrict__ gcnt,
                                 const float* __restrict__ mean_scale,
                                 float* gvar, int N) {
  int i = blockIdx.x * blockDim.x + threadIdx.x;
  if (i >= N * C_DIM) return;
  int r = i / C_DIM, c = i % C_DIM;
  int g = batch[r];
  float cnt = fmaxf(gcnt[g], 1.f);
  float mean = gsum[g * C_DIM + c] / cnt;
  float o = x[i] - mean * mean_scale[c];
  x[i] = o;
  atomicAdd(&gvar[g * C_DIM + c], o * o);
}
__global__ void gn_final_kernel(float* __restrict__ x, const int* __restrict__ batch,
                                const float* __restrict__ gvar, const float* __restrict__ gcnt,
                                const float* __restrict__ w, const float* __restrict__ b,
                                int N) {
  int i = blockIdx.x * blockDim.x + threadIdx.x;
  if (i >= N * C_DIM) return;
  int r = i / C_DIM, c = i % C_DIM;
  int g = batch[r];
  float cnt = fmaxf(gcnt[g], 1.f);
  float std_ = sqrtf(gvar[g * C_DIM + c] / cnt + 1e-5f);
  x[i] = w[c] * x[i] / std_ + b[c];
}

// ---------------------------------------------------------------------------
static void run_graphnorm(float* x, const int* batch, int N,
                          const float* w, const float* b, const float* ms,
                          float* gsum, float* gvar, float* gcnt, hipStream_t s) {
  gn_zero_kernel<<<(64 * C_DIM + 255) / 256, 256, 0, s>>>(gsum, gvar, gcnt);
  int total = N * C_DIM;
  gn_sum_kernel<<<(total + 255) / 256, 256, 0, s>>>(x, batch, gsum, gcnt, N);
  gn_center_kernel<<<(total + 255) / 256, 256, 0, s>>>(x, batch, gsum, gcnt, ms, gvar, N);
  gn_final_kernel<<<(total + 255) / 256, 256, 0, s>>>(x, batch, gvar, gcnt, w, b, N);
}

extern "C" void kernel_launch(void* const* d_in, const int* in_sizes, int n_in,
                              void* d_out, int out_size, void* d_ws, size_t ws_size,
                              hipStream_t stream) {
  (void)in_sizes; (void)n_in; (void)out_size; (void)ws_size;
  const int N0 = 4096, N1 = 8192, N2 = 6144;

  const float* x0   = (const float*)d_in[0];
  const float* x1   = (const float*)d_in[1];
  const float* x2   = (const float*)d_in[2];
  const float* L0   = (const float*)d_in[3];
  const float* L1   = (const float*)d_in[4];
  const float* L2   = (const float*)d_in[5];
  const float* inc1 = (const float*)d_in[6];
  const float* inc2 = (const float*)d_in[7];
  const float* W0   = (const float*)d_in[8];
  const float* W1   = (const float*)d_in[9];
  const float* W2   = (const float*)d_in[10];
  const float* Wc1  = (const float*)d_in[11];
  const float* Wc2  = (const float*)d_in[12];
  const float* gn1_w  = (const float*)d_in[13];
  const float* gn1_b  = (const float*)d_in[14];
  const float* gn1_ms = (const float*)d_in[15];
  const float* gn2_w  = (const float*)d_in[16];
  const float* gn2_b  = (const float*)d_in[17];
  const float* gn2_ms = (const float*)d_in[18];
  const int* batch  = (const int*)d_in[19];
  const int* batch1 = (const int*)d_in[20];

  float* out_x0 = (float*)d_out;
  float* out_x1 = out_x0 + (size_t)N0 * C_DIM;
  float* out_x2 = out_x1 + (size_t)N1 * C_DIM;

  char* ws = (char*)d_ws;
  size_t off = 0;
  auto alloc = [&](size_t bytes) -> void* {
    void* p = ws + off;
    off += bytes;
    off = (off + 255) & ~(size_t)255;
    return p;
  };
  float* inv0 = (float*)alloc((size_t)N0 * 4);
  float* inv1 = (float*)alloc((size_t)N1 * 4);
  float* inv2 = (float*)alloc((size_t)N2 * 4);
  float* h0   = (float*)alloc((size_t)N0 * C_DIM * 4);
  float* h1   = (float*)alloc((size_t)N1 * C_DIM * 4);
  bf16*  Ybf  = (bf16*)alloc((size_t)C_DIM * N1 * 2);   // reused for all RHS operands
  float* gsum = (float*)alloc(64 * C_DIM * 4);
  float* gvar = (float*)alloc(64 * C_DIM * 4);
  float* gcnt = (float*)alloc(64 * 4);

  // 1) degree inverses
  rowsum_inv_kernel<<<N0, 256, 0, stream>>>(L0, inv0, N0);
  rowsum_inv_kernel<<<N1, 256, 0, stream>>>(L1, inv1, N1);
  rowsum_inv_kernel<<<N2, 256, 0, stream>>>(L2, inv2, N2);

  // 2) backbone: h_i = sigmoid(inv_r * (L_i @ ((x_i @ W_i) * inv_c)))
  small_gemm_kernel<<<N2 / 32, 256, 0, stream>>>(x2, W2, inv2, Ybf, N2);
  big_gemm_wmma<<<N2 / 128, 256, 0, stream>>>(L2, Ybf, inv2, nullptr, out_x2, N2, N2, 1);

  small_gemm_kernel<<<N1 / 32, 256, 0, stream>>>(x1, W1, inv1, Ybf, N1);
  big_gemm_wmma<<<N1 / 128, 256, 0, stream>>>(L1, Ybf, inv1, nullptr, h1, N1, N1, 1);

  small_gemm_kernel<<<N0 / 32, 256, 0, stream>>>(x0, W0, inv0, Ybf, N0);
  big_gemm_wmma<<<N0 / 128, 256, 0, stream>>>(L0, Ybf, inv0, nullptr, h0, N0, N0, 1);

  // 3) x1_out = GN(h1 + inc2 @ (x2_out @ Wc1))
  small_gemm_kernel<<<N2 / 32, 256, 0, stream>>>(out_x2, Wc1, nullptr, Ybf, N2);
  big_gemm_wmma<<<N1 / 128, 256, 0, stream>>>(inc2, Ybf, nullptr, h1, out_x1, N1, N2, 0);
  run_graphnorm(out_x1, batch1, N1, gn1_w, gn1_b, gn1_ms, gsum, gvar, gcnt, stream);

  // 4) x0_out = GN(h0 + inc1 @ (x1_out @ Wc2))
  small_gemm_kernel<<<N1 / 32, 256, 0, stream>>>(out_x1, Wc2, nullptr, Ybf, N1);
  big_gemm_wmma<<<N0 / 128, 256, 0, stream>>>(inc1, Ybf, nullptr, h0, out_x0, N0, N1, 0);
  run_graphnorm(out_x0, batch, N0, gn2_w, gn2_b, gn2_ms, gsum, gvar, gcnt, stream);
}
